// ArtemisSubModule_20504173871772
// MI455X (gfx1250) — compile-verified, matrix operand order checked by reading, not run on hardware
//
#include <hip/hip_runtime.h>
#include <hip/hip_bf16.h>

// ---- problem constants (from reference) ----
#define VOCAB  50257
#define E_DIM  128
#define DE_DIM 2048
#define DD_DIM 512
#define A_DIM  512
#define AUX_N  128
#define P_N    196
#define KBEAM  5
#define SOS_ID 1
#define UNK_ID 2
#define NSTEPS 20
#define INV_TEMP 2.0f   // 1 / 0.5

typedef __attribute__((ext_vector_type(16))) _Float16 v16h;
typedef __attribute__((ext_vector_type(8)))  float    v8f;

// ---------------------------------------------------------------------------
// Generic WMMA GEMM:  C[M x N] = scale * (A[M x Kd] @ op(B) + bias) (+ C if ACC)
// op(B): BT==0 -> B is [Kd x N] row-major (ldb = row stride)
//        BT==1 -> B is [N x Kd] row-major (ldb = row stride), i.e. uses B^T
// Kd must be a multiple of 32; lda/ldb(BT=1) must be even (float2 loads).
// One wave computes one 16x16 output tile with v_wmma_f32_16x16x32_f16.
// Out-of-range rows/cols are index-CLAMPED (not predicated): garbage lands
// only in D rows/cols that the guarded store discards, so the inner loop has
// zero divergence and loads batch under pipelined s_wait_loadcnt.
// BT/ACC are compile-time so the dead B-path is fully removed.
// ---------------------------------------------------------------------------
template <int BT, int ACC>
__global__ __launch_bounds__(128)
void wmma_gemm_kernel(const float* __restrict__ A, int lda,
                      const float* __restrict__ B, int ldb,
                      const float* __restrict__ bias,
                      float* __restrict__ C, int ldc,
                      int M, int N, int Kd, float scale)
{
    const int lane = threadIdx.x & 31;
    const int wave = threadIdx.x >> 5;
    const int wavesPerBlock = blockDim.x >> 5;
    const int ntile = blockIdx.x * wavesPerBlock + wave;
    const int mtile = blockIdx.y;
    const int n0 = ntile * 16;
    const int m0 = mtile * 16;
    if (n0 >= N) return;                    // wave-uniform exit

    const int mlo = lane & 15;              // A: M index ; B/C: N index
    const int hi  = lane >> 4;              // which K-half this lane holds

    const int mA = min(m0 + mlo, M - 1);    // clamped row of A
    const int nB = min(n0 + mlo, N - 1);    // clamped col of B

    const float* __restrict__ Arow = A + (size_t)mA * lda;
    const float* __restrict__ Bt   = B + (size_t)nB * ldb;   // BT==1 path
    const float* __restrict__ Bn   = B + nB;                 // BT==0 path

    v8f acc = {};
#pragma unroll 2
    for (int kk = 0; kk < Kd; kk += 32) {
        if (!BT) {
            // branch-free prefetch of the next K-slab (clamped at the tail);
            // locality=3 -> near-temporal, keep next_W resident in L2
            const int kp = (kk + 32 < Kd) ? (kk + 32) : kk;
            __builtin_prefetch(Bn + (size_t)kp * ldb, 0, 3);
        }
        v16h a, b;
#pragma unroll
        for (int v = 0; v < 8; ++v) {
            const int kb = kk + (v < 4 ? 2 * v : 16 + 2 * (v - 4)) + 8 * hi;
            const float2 a2 = *(const float2*)(Arow + kb);   // even offset -> 8B aligned
            a[2 * v]     = (_Float16)a2.x;
            a[2 * v + 1] = (_Float16)a2.y;
            float y0, y1;
            if (BT) {
                const float2 b2 = *(const float2*)(Bt + kb);
                y0 = b2.x; y1 = b2.y;
            } else {
                y0 = Bn[(size_t)kb * ldb];
                y1 = Bn[(size_t)(kb + 1) * ldb];
            }
            b[2 * v]     = (_Float16)y0;
            b[2 * v + 1] = (_Float16)y1;
        }
        acc = __builtin_amdgcn_wmma_f32_16x16x32_f16(
            /*neg_a=*/false, a, /*neg_b=*/false, b,
            /*c_mod=*/(short)0, acc, /*reuse_a=*/false, /*reuse_b=*/false);
    }
    // ---- store: lane's column n, rows m = v + 8*hi
    const int n = n0 + mlo;
    if (n < N) {
        const float bv = bias[n];
#pragma unroll
        for (int v = 0; v < 8; ++v) {
            const int m = m0 + v + 8 * hi;
            if (m < M) {
                const size_t off = (size_t)m * ldc + n;
                float r = (acc[v] + bv) * scale;
                if (ACC) r += C[off];
                C[off] = r;
            }
        }
    }
}

// ---------------------------------------------------------------------------
// mean over P of encoder_out -> mean_enc [K x DE]
// ---------------------------------------------------------------------------
__global__ void mean_kernel(const float* __restrict__ encoder_out,
                            float* __restrict__ mean_enc)
{
    int idx = blockIdx.x * blockDim.x + threadIdx.x;   // over K*DE
    if (idx >= KBEAM * DE_DIM) return;
    int k = idx / DE_DIM, d = idx % DE_DIM;
    const float* eo = encoder_out + (size_t)k * P_N * DE_DIM + d;
    float s = 0.0f;
    for (int p = 0; p < P_N; ++p) s += eo[(size_t)p * DE_DIM];
    mean_enc[idx] = s * (1.0f / (float)P_N);
}

__global__ void init_state_kernel(int* __restrict__ prev_words,
                                  float* __restrict__ tks)
{
    int t = threadIdx.x;
    if (t < KBEAM) { prev_words[t] = SOS_ID; tks[t] = 0.0f; }
}

// ---------------------------------------------------------------------------
// Attention + decoder-input build.  One block (256 thr, 8 waves) per beam.
//   e[p]   = sum_a tanh(enc_att[k,p,a] + attdec[k,a]) * wfull[a] + bfull
//   alpha  = softmax_p(e)
//   awe[d] = sum_p alpha[p] * encoder_out[k,p,d]
//   x      = [ emb(prev) | sigmoid(fbeta)*awe | aux ]
// ---------------------------------------------------------------------------
__global__ __launch_bounds__(256)
void attention_kernel(const float* __restrict__ enc_att,
                      const float* __restrict__ encoder_out,
                      const float* __restrict__ attdec,
                      const float* __restrict__ wfull,
                      const float* __restrict__ bfull_p,
                      const float* __restrict__ fbeta,
                      const float* __restrict__ W_emb,
                      const float* __restrict__ aux_feat,
                      const int*   __restrict__ prev_words,
                      float* __restrict__ x)
{
    const int k = blockIdx.x;
    const int t = threadIdx.x;
    const int lane = t & 31, wave = t >> 5;

    __shared__ float s_ad[A_DIM];
    __shared__ float s_e[P_N];
    __shared__ float s_red[8];
    __shared__ float s_red2[8];

    for (int i = t; i < A_DIM; i += 256) s_ad[i] = attdec[k * A_DIM + i];
    __syncthreads();

    const float bfull = bfull_p[0];
    for (int p = wave; p < P_N; p += 8) {
        const float* ea = enc_att + ((size_t)k * P_N + p) * A_DIM;
        float sum = 0.0f;
        for (int a = lane; a < A_DIM; a += 32)
            sum += tanhf(ea[a] + s_ad[a]) * wfull[a];
        for (int o = 16; o; o >>= 1) sum += __shfl_xor(sum, o, 32);
        if (lane == 0) s_e[p] = sum + bfull;
    }
    __syncthreads();

    // softmax over P
    float lm = -1e30f;
    for (int p = t; p < P_N; p += 256) lm = fmaxf(lm, s_e[p]);
    for (int o = 16; o; o >>= 1) lm = fmaxf(lm, __shfl_xor(lm, o, 32));
    if (lane == 0) s_red[wave] = lm;
    __syncthreads();
    float mx = s_red[0];
    for (int w = 1; w < 8; ++w) mx = fmaxf(mx, s_red[w]);

    float ls = 0.0f;
    for (int p = t; p < P_N; p += 256) ls += __expf(s_e[p] - mx);
    for (int o = 16; o; o >>= 1) ls += __shfl_xor(ls, o, 32);
    if (lane == 0) s_red2[wave] = ls;
    __syncthreads();
    float ssum = 0.0f;
    for (int w = 0; w < 8; ++w) ssum += s_red2[w];
    const float inv = 1.0f / ssum;
    __syncthreads();
    for (int p = t; p < P_N; p += 256) s_e[p] = __expf(s_e[p] - mx) * inv;
    __syncthreads();

    // awe + gated write
    const float* eo = encoder_out + (size_t)k * P_N * DE_DIM;
    float* xr = x + (size_t)k * (E_DIM + DE_DIM + AUX_N);
    for (int d = t; d < DE_DIM; d += 256) {
        float awe = 0.0f;
        for (int p = 0; p < P_N; ++p) awe += s_e[p] * eo[(size_t)p * DE_DIM + d];
        float g = 1.0f / (1.0f + __expf(-fbeta[k * DE_DIM + d]));
        xr[E_DIM + d] = g * awe;
    }
    const int w = prev_words[k];
    for (int i = t; i < E_DIM; i += 256) {
        xr[i] = W_emb[(size_t)w * E_DIM + i];
        xr[E_DIM + DE_DIM + i] = aux_feat[i];
    }
}

// ---------------------------------------------------------------------------
// LSTM pointwise (gates are torch i,f,g,o order, 4*DD per beam)
// ---------------------------------------------------------------------------
__global__ void lstm_pointwise(const float* __restrict__ gates,
                               const float* __restrict__ c,
                               float* __restrict__ h_new,
                               float* __restrict__ c_new)
{
    int idx = blockIdx.x * blockDim.x + threadIdx.x;   // K*DD
    if (idx >= KBEAM * DD_DIM) return;
    int k = idx / DD_DIM, j = idx % DD_DIM;
    const float* g = gates + (size_t)k * 4 * DD_DIM;
    float gi = g[j], gf = g[DD_DIM + j], gg = g[2 * DD_DIM + j], go = g[3 * DD_DIM + j];
    float si = 1.0f / (1.0f + __expf(-gi));
    float sf = 1.0f / (1.0f + __expf(-gf));
    float so = 1.0f / (1.0f + __expf(-go));
    float cn = sf * c[idx] + si * tanhf(gg);
    c_new[idx] = cn;
    h_new[idx] = so * tanhf(cn);
}

// ---------------------------------------------------------------------------
// Per-beam max and log-sum-exp over V (for log_softmax)
// ---------------------------------------------------------------------------
__global__ __launch_bounds__(256)
void logsumexp_kernel(const float* __restrict__ scores,
                      float* __restrict__ redm, float* __restrict__ redl)
{
    const int k = blockIdx.x, t = threadIdx.x;
    const int lane = t & 31, wave = t >> 5;
    const float* s = scores + (size_t)k * VOCAB;
    __shared__ float sm[8], ss[8];

    float m = -1e30f;
    for (int v = t; v < VOCAB; v += 256) m = fmaxf(m, s[v]);
    for (int o = 16; o; o >>= 1) m = fmaxf(m, __shfl_xor(m, o, 32));
    if (lane == 0) sm[wave] = m;
    __syncthreads();
    float M = sm[0];
    for (int w = 1; w < 8; ++w) M = fmaxf(M, sm[w]);

    float sum = 0.0f;
    for (int v = t; v < VOCAB; v += 256) sum += __expf(s[v] - M);
    for (int o = 16; o; o >>= 1) sum += __shfl_xor(sum, o, 32);
    if (lane == 0) ss[wave] = sum;
    __syncthreads();
    if (t == 0) {
        float S = 0.0f;
        for (int w = 0; w < 8; ++w) S += ss[w];
        redm[k] = M;
        redl[k] = logf(S);
    }
}

// ---------------------------------------------------------------------------
// Top-5 selection + beam gather + output writes. Single block, 256 threads.
// ---------------------------------------------------------------------------
__device__ inline void insert5(float (&bv)[5], long (&bi)[5], float v, long i)
{
    if (v <= bv[4]) return;
    bv[4] = v; bi[4] = i;
#pragma unroll
    for (int j = 4; j > 0; --j) {
        if (bv[j] > bv[j - 1]) {
            float tv = bv[j]; bv[j] = bv[j - 1]; bv[j - 1] = tv;
            long  ti = bi[j]; bi[j] = bi[j - 1]; bi[j - 1] = ti;
        }
    }
}

__global__ __launch_bounds__(256)
void topk_kernel(const float* __restrict__ scores,
                 const float* __restrict__ redm, const float* __restrict__ redl,
                 float* __restrict__ tks, int* __restrict__ prev_words,
                 const float* __restrict__ h_new, const float* __restrict__ c_new,
                 float* __restrict__ h, float* __restrict__ c,
                 int step, float* __restrict__ out)
{
    const int t = threadIdx.x;
    __shared__ float cv[256 * 5];
    __shared__ long  ci[256 * 5];
    __shared__ float topv[5];
    __shared__ long  topi[5];
    __shared__ int   pis[5];

    float bv[5]; long bi[5];
#pragma unroll
    for (int i = 0; i < 5; ++i) { bv[i] = -1e30f; bi[i] = -1; }

    if (step == 1) {
        const float m = redm[0], l = redl[0];
        for (int v = t; v < VOCAB; v += 256) {
            float sc = (v == UNK_ID) ? -1e30f : (scores[v] - m - l);
            insert5(bv, bi, sc, (long)v);
        }
    } else {
        for (long idx = t; idx < (long)KBEAM * VOCAB; idx += 256) {
            int k = (int)(idx / VOCAB), v = (int)(idx % VOCAB);
            float sc = (v == UNK_ID)
                     ? -1e30f
                     : (tks[k] + scores[(size_t)k * VOCAB + v] - redm[k] - redl[k]);
            insert5(bv, bi, sc, idx);
        }
    }
#pragma unroll
    for (int i = 0; i < 5; ++i) { cv[t * 5 + i] = bv[i]; ci[t * 5 + i] = bi[i]; }
    __syncthreads();

    if (t == 0) {
        for (int s = 0; s < 5; ++s) {
            float best = -1e31f; int bj = 0;
            for (int j = 0; j < 256 * 5; ++j)
                if (ci[j] >= 0 && cv[j] > best) { best = cv[j]; bj = j; }
            topv[s] = best; topi[s] = ci[bj]; ci[bj] = -1;
        }
    }
    __syncthreads();

    if (t < KBEAM) {
        long fi = topi[t];
        int p, n;
        if (step == 1) { p = 0; n = (int)fi; }
        else           { p = (int)(fi / VOCAB); n = (int)(fi % VOCAB); }
        pis[t] = p;
        tks[t] = topv[t];
        prev_words[t] = n;
        // output layout: seq[NSTEPS*K] | prev_inds[(NSTEPS-1)*K] | tks[K]
        out[(step - 1) * KBEAM + t] = (float)n;
        if (step >= 2)      out[NSTEPS * KBEAM + (step - 2) * KBEAM + t] = (float)p;
        if (step == NSTEPS) out[NSTEPS * KBEAM + (NSTEPS - 1) * KBEAM + t] = topv[t];
    }
    __syncthreads();

    for (int idx = t; idx < KBEAM * DD_DIM; idx += 256) {
        int k = idx / DD_DIM, j = idx % DD_DIM;
        h[idx] = h_new[pis[k] * DD_DIM + j];
        c[idx] = c_new[pis[k] * DD_DIM + j];
    }
}

// ---------------------------------------------------------------------------
// host side
// ---------------------------------------------------------------------------
static void launch_gemm(hipStream_t stream, const float* A, int lda,
                        const float* B, int ldb, int bT, const float* bias,
                        float* C, int ldc, int M, int N, int Kd,
                        float scale, int accum)
{
    const int wpb = 4;
    int ntiles = (N + 15) / 16;
    int mtiles = (M + 15) / 16;
    dim3 grid((ntiles + wpb - 1) / wpb, mtiles);
    dim3 block(32 * wpb);
    if (bT) {
        if (accum)
            hipLaunchKernelGGL((wmma_gemm_kernel<1, 1>), grid, block, 0, stream,
                               A, lda, B, ldb, bias, C, ldc, M, N, Kd, scale);
        else
            hipLaunchKernelGGL((wmma_gemm_kernel<1, 0>), grid, block, 0, stream,
                               A, lda, B, ldb, bias, C, ldc, M, N, Kd, scale);
    } else {
        if (accum)
            hipLaunchKernelGGL((wmma_gemm_kernel<0, 1>), grid, block, 0, stream,
                               A, lda, B, ldb, bias, C, ldc, M, N, Kd, scale);
        else
            hipLaunchKernelGGL((wmma_gemm_kernel<0, 0>), grid, block, 0, stream,
                               A, lda, B, ldb, bias, C, ldc, M, N, Kd, scale);
    }
}

extern "C" void kernel_launch(void* const* d_in, const int* in_sizes, int n_in,
                              void* d_out, int out_size, void* d_ws, size_t ws_size,
                              hipStream_t stream)
{
    const float* encoder_out = (const float*)d_in[0];
    const float* aux_feat    = (const float*)d_in[1];
    const float* W_emb       = (const float*)d_in[2];
    const float* att_enc_W   = (const float*)d_in[3];
    const float* att_enc_b   = (const float*)d_in[4];
    const float* att_dec_W   = (const float*)d_in[5];
    const float* att_dec_b   = (const float*)d_in[6];
    const float* att_full_W  = (const float*)d_in[7];
    const float* att_full_b  = (const float*)d_in[8];
    const float* f_beta_W    = (const float*)d_in[9];
    const float* f_beta_b    = (const float*)d_in[10];
    const float* init_h_W    = (const float*)d_in[11];
    const float* init_h_b    = (const float*)d_in[12];
    const float* init_c_W    = (const float*)d_in[13];
    const float* init_c_b    = (const float*)d_in[14];
    const float* lstm_Wih    = (const float*)d_in[15];
    const float* lstm_Whh    = (const float*)d_in[16];
    const float* lstm_bih    = (const float*)d_in[17];
    const float* lstm_bhh    = (const float*)d_in[18];
    const float* next_W      = (const float*)d_in[19];
    const float* next_b      = (const float*)d_in[20];
    // d_in[21] = n_steps (hardcoded NSTEPS=20: graph capture needs a fixed launch count)

    float* ws  = (float*)d_ws;
    size_t off = 0;
    auto carve = [&](size_t n) -> float* {
        float* p = ws + off;
        off += (n + 63) & ~(size_t)63;
        return p;
    };
    const int XDIM = E_DIM + DE_DIM + AUX_N;   // 2304
    float* enc_att  = carve((size_t)KBEAM * P_N * A_DIM);  // [K*P x A]
    float* h        = carve(KBEAM * DD_DIM);
    float* c        = carve(KBEAM * DD_DIM);
    float* h_new    = carve(KBEAM * DD_DIM);
    float* c_new    = carve(KBEAM * DD_DIM);
    float* mean_enc = carve(KBEAM * DE_DIM);
    float* attdec   = carve(KBEAM * A_DIM);
    float* fbeta    = carve(KBEAM * DE_DIM);
    float* xbuf     = carve(KBEAM * XDIM);
    float* gates    = carve(KBEAM * 4 * DD_DIM);
    float* scores   = carve((size_t)KBEAM * VOCAB);
    float* redm     = carve(KBEAM);
    float* redl     = carve(KBEAM);
    float* tks      = carve(KBEAM);
    int*   prevw    = (int*)carve(KBEAM);
    (void)ws_size; (void)in_sizes; (void)n_in; (void)out_size;

    // ---- one-time precompute ----
    // enc_att = encoder_out @ att_enc_W + b   : [980 x 2048] @ [2048 x 512]
    launch_gemm(stream, encoder_out, DE_DIM, att_enc_W, A_DIM, 0, att_enc_b,
                enc_att, A_DIM, KBEAM * P_N, A_DIM, DE_DIM, 1.0f, 0);
    mean_kernel<<<(KBEAM * DE_DIM + 255) / 256, 256, 0, stream>>>(encoder_out, mean_enc);
    launch_gemm(stream, mean_enc, DE_DIM, init_h_W, DD_DIM, 0, init_h_b,
                h, DD_DIM, KBEAM, DD_DIM, DE_DIM, 1.0f, 0);
    launch_gemm(stream, mean_enc, DE_DIM, init_c_W, DD_DIM, 0, init_c_b,
                c, DD_DIM, KBEAM, DD_DIM, DE_DIM, 1.0f, 0);
    init_state_kernel<<<1, 32, 0, stream>>>(prevw, tks);

    // ---- beam-search steps ----
    for (int step = 1; step <= NSTEPS; ++step) {
        // attdec = h @ att_dec_W + b          : [5x512]@[512x512]
        launch_gemm(stream, h, DD_DIM, att_dec_W, A_DIM, 0, att_dec_b,
                    attdec, A_DIM, KBEAM, A_DIM, DD_DIM, 1.0f, 0);
        // fbeta = h @ f_beta_W + b            : [5x512]@[512x2048]
        launch_gemm(stream, h, DD_DIM, f_beta_W, DE_DIM, 0, f_beta_b,
                    fbeta, DE_DIM, KBEAM, DE_DIM, DD_DIM, 1.0f, 0);
        attention_kernel<<<KBEAM, 256, 0, stream>>>(
            enc_att, encoder_out, attdec, att_full_W, att_full_b,
            fbeta, W_emb, aux_feat, prevw, xbuf);
        // gates = x @ Wih^T + bih             : [5x2304]@[2304x2048]
        launch_gemm(stream, xbuf, XDIM, lstm_Wih, XDIM, 1, lstm_bih,
                    gates, 4 * DD_DIM, KBEAM, 4 * DD_DIM, XDIM, 1.0f, 0);
        // gates += h @ Whh^T + bhh            : [5x512]@[512x2048]
        launch_gemm(stream, h, DD_DIM, lstm_Whh, DD_DIM, 1, lstm_bhh,
                    gates, 4 * DD_DIM, KBEAM, 4 * DD_DIM, DD_DIM, 1.0f, 1);
        lstm_pointwise<<<(KBEAM * DD_DIM + 255) / 256, 256, 0, stream>>>(
            gates, c, h_new, c_new);
        // logits/TEMP = (h_new @ next_W + b) * 2 : [5x512]@[512x50257]  (the 103MB/step GEMM)
        launch_gemm(stream, h_new, DD_DIM, next_W, VOCAB, 0, next_b,
                    scores, VOCAB, KBEAM, VOCAB, DD_DIM, INV_TEMP, 0);
        logsumexp_kernel<<<KBEAM, 256, 0, stream>>>(scores, redm, redl);
        topk_kernel<<<1, 256, 0, stream>>>(scores, redm, redl, tks, prevw,
                                           h_new, c_new, h, c, step, (float*)d_out);
    }
}
